// Router_8572754723466
// MI455X (gfx1250) — matrix operand order repaired and checked by reading, not compiled, not run on hardware
//
#include <hip/hip_runtime.h>

#define NROWS 8192
#define DIN   2048
#define DOUT  2048
#define NEXP  8

#define BM 64
#define BN 128
#define BK 32
#define MT (NROWS / BM)   // 128 worst-case m-tiles per expert
#define LDA 40            // padded LDS row stride (bf16 elems), 80B = 16B-aligned

typedef __attribute__((ext_vector_type(16))) __bf16 v16bf;
typedef __attribute__((ext_vector_type(8)))  __bf16 v8bf;
typedef __attribute__((ext_vector_type(8)))  float  v8f;

// ---------------------------------------------------------------- init
__global__ void zero_counts_kernel(int* __restrict__ counts) {
  if (threadIdx.x < NEXP) counts[threadIdx.x] = 0;
}

// ---------------------------------------------------------------- router
// one wave32 per row: 8 dot products of length DIN, shuffle-reduce, argmax,
// append row to its expert's list.
__global__ __launch_bounds__(256) void router_kernel(
    const float* __restrict__ cond, const float* __restrict__ Wp,
    const float* __restrict__ bp, int* __restrict__ counts,
    int* __restrict__ rowlist)
{
  const int lane = threadIdx.x & 31;
  const int wv   = threadIdx.x >> 5;
  const int n    = blockIdx.x * 8 + wv;
  if (n >= NROWS) return;

  const float* c = cond + (size_t)n * DIN;
  float acc[NEXP];
#pragma unroll
  for (int e = 0; e < NEXP; ++e) acc[e] = 0.f;

  for (int d = lane; d < DIN; d += 32) {
    const float x = c[d];
#pragma unroll
    for (int e = 0; e < NEXP; ++e) acc[e] += x * Wp[e * DIN + d];
  }
#pragma unroll
  for (int off = 16; off >= 1; off >>= 1) {
#pragma unroll
    for (int e = 0; e < NEXP; ++e) acc[e] += __shfl_down(acc[e], off, 32);
  }
  if (lane == 0) {
    float best = acc[0] + bp[0];
    int   bi   = 0;
#pragma unroll
    for (int e = 1; e < NEXP; ++e) {
      const float v = acc[e] + bp[e];
      if (v > best) { best = v; bi = e; }   // strict '>' == first-max (jnp.argmax)
    }
    const int pos = atomicAdd(&counts[bi], 1);
    rowlist[bi * NROWS + pos] = n;
  }
}

// ---------------------------------------------------------------- helpers
__device__ __forceinline__ v8bf cvt8(const float4 a, const float4 b) {
  v8bf v;
  v[0] = (__bf16)a.x; v[1] = (__bf16)a.y; v[2] = (__bf16)a.z; v[3] = (__bf16)a.w;
  v[4] = (__bf16)b.x; v[5] = (__bf16)b.y; v[6] = (__bf16)b.z; v[7] = (__bf16)b.w;
  return v;
}

__device__ __forceinline__ v16bf frag_join(const __bf16* lo_p, const __bf16* hi_p) {
  const v8bf lo = *(const v8bf*)lo_p;
  const v8bf hi = *(const v8bf*)hi_p;
  v16bf r;
#pragma unroll
  for (int i = 0; i < 8; ++i) { r[i] = lo[i]; r[i + 8] = hi[i]; }
  return r;
}

// ---------------------------------------------------------------- grouped GEMM
// block: 256 threads = 8 waves; tile: BM x BN, K in chunks of 32 (bf16 WMMA).
// Double-buffered LDS, one barrier per K-chunk. sched_barriers pin the
// phase order: [global loads k+1] | [LDS frags + WMMA k] | [cvt + LDS store],
// so the global loads stay in flight across the whole WMMA section.
__global__ __launch_bounds__(256) void expert_gemm_kernel(
    const float* __restrict__ X, const float* __restrict__ We,
    const float* __restrict__ be, const int* __restrict__ counts,
    const int* __restrict__ rowlist, float* __restrict__ out)
{
  __shared__ __align__(16) __bf16 sA[2][BM * LDA];
  __shared__ __align__(16) __bf16 sB[2][BN * LDA];
  __shared__ int srow[BM];

  const int e  = blockIdx.y / MT;
  const int mt = blockIdx.y % MT;
  const int cnt = counts[e];
  if (mt * BM >= cnt) return;                       // early exit: empty tile
  int cnt_local = cnt - mt * BM;
  if (cnt_local > BM) cnt_local = BM;

  const int tid = threadIdx.x;
  if (tid < BM) {
    const int i = (tid < cnt_local) ? tid : (cnt_local - 1);  // clamp padding
    srow[tid] = rowlist[e * NROWS + mt * BM + i];
  }
  __syncthreads();

  const int lane  = tid & 31;
  const int wv    = tid >> 5;
  const int nlane = lane & 15;
  const int fbase = blockIdx.x * BN;
  const int fcol  = fbase + wv * 16 + nlane;        // this lane's output column

  // A staging: thread -> (row 0..63, 8 floats)
  const int arow = tid >> 2;
  const int aoff = (tid & 3) * 8;
  const float* aptr = X + (size_t)srow[arow] * DIN + aoff;
  // B staging: thread -> (row 0..127, 16 floats), coalesced along k
  const int brow = tid >> 1;
  const int boff = (tid & 1) * 16;
  const float* bptr = We + ((size_t)e * DOUT + (size_t)(fbase + brow)) * DIN + boff;

  // fragment k-bases per ISA VGPR layouts
  const int akb = (lane < 16) ? 0 : 8;    // A: split pattern {kb..kb+7, kb+16..kb+23}
  const int bkb = (lane < 16) ? 0 : 8;    // B: split pattern {kb..kb+7, kb+16..kb+23}

  const int aIdx0 = (0 * 16 + nlane) * LDA + akb;
  const int aIdx1 = (1 * 16 + nlane) * LDA + akb;
  const int aIdx2 = (2 * 16 + nlane) * LDA + akb;
  const int aIdx3 = (3 * 16 + nlane) * LDA + akb;
  const int bIdx  = (wv * 16 + nlane) * LDA + bkb;

  const int aSt = arow * LDA + aoff;
  const int bSt = brow * LDA + boff;

  v8f acc0 = {}, acc1 = {}, acc2 = {}, acc3 = {};

  // ---- prologue: stage chunk 0 into buffer 0
  {
    const float4 aA = *(const float4*)(aptr);
    const float4 aB = *(const float4*)(aptr + 4);
    const float4 b0 = *(const float4*)(bptr);
    const float4 b1 = *(const float4*)(bptr + 4);
    const float4 b2 = *(const float4*)(bptr + 8);
    const float4 b3 = *(const float4*)(bptr + 12);
    *(v8bf*)&sA[0][aSt]     = cvt8(aA, aB);
    *(v8bf*)&sB[0][bSt]     = cvt8(b0, b1);
    *(v8bf*)&sB[0][bSt + 8] = cvt8(b2, b3);
  }
  __syncthreads();

  int p = 0;
  // steady state: load chunk k0, compute chunk k0-BK
  for (int k0 = BK; k0 < DIN; k0 += BK) {
    // ---- phase 1: issue next chunk's global loads
    const float4 nA0 = *(const float4*)(aptr + k0);
    const float4 nA1 = *(const float4*)(aptr + k0 + 4);
    const float4 nB0 = *(const float4*)(bptr + k0);
    const float4 nB1 = *(const float4*)(bptr + k0 + 4);
    const float4 nB2 = *(const float4*)(bptr + k0 + 8);
    const float4 nB3 = *(const float4*)(bptr + k0 + 12);
    __builtin_amdgcn_sched_barrier(0);   // loads may not sink below

    // ---- phase 2: compute current chunk from buffer p (loads in flight)
    const __bf16* A = sA[p];
    const __bf16* B = sB[p];
    const v16bf bf  = frag_join(&B[bIdx], &B[bIdx] + 16);
    const v16bf a0f = frag_join(&A[aIdx0], &A[aIdx0] + 16);
    const v16bf a1f = frag_join(&A[aIdx1], &A[aIdx1] + 16);
    const v16bf a2f = frag_join(&A[aIdx2], &A[aIdx2] + 16);
    const v16bf a3f = frag_join(&A[aIdx3], &A[aIdx3] + 16);

    acc0 = __builtin_amdgcn_wmma_f32_16x16x32_bf16(false, a0f, false, bf, (short)0, acc0, false, false);
    acc1 = __builtin_amdgcn_wmma_f32_16x16x32_bf16(false, a1f, false, bf, (short)0, acc1, false, false);
    acc2 = __builtin_amdgcn_wmma_f32_16x16x32_bf16(false, a2f, false, bf, (short)0, acc2, false, false);
    acc3 = __builtin_amdgcn_wmma_f32_16x16x32_bf16(false, a3f, false, bf, (short)0, acc3, false, false);
    __builtin_amdgcn_sched_barrier(0);   // waits/cvt may not hoist above

    // ---- phase 3: convert + stage next chunk into the other buffer
    const int q = p ^ 1;
    *(v8bf*)&sA[q][aSt]     = cvt8(nA0, nA1);
    *(v8bf*)&sB[q][bSt]     = cvt8(nB0, nB1);
    *(v8bf*)&sB[q][bSt + 8] = cvt8(nB2, nB3);

    __syncthreads();
    p = q;
  }

  // ---- epilogue chunk: compute last staged buffer
  {
    const __bf16* A = sA[p];
    const __bf16* B = sB[p];
    const v16bf bf  = frag_join(&B[bIdx], &B[bIdx] + 16);
    const v16bf a0f = frag_join(&A[aIdx0], &A[aIdx0] + 16);
    const v16bf a1f = frag_join(&A[aIdx1], &A[aIdx1] + 16);
    const v16bf a2f = frag_join(&A[aIdx2], &A[aIdx2] + 16);
    const v16bf a3f = frag_join(&A[aIdx3], &A[aIdx3] + 16);

    acc0 = __builtin_amdgcn_wmma_f32_16x16x32_bf16(false, a0f, false, bf, (short)0, acc0, false, false);
    acc1 = __builtin_amdgcn_wmma_f32_16x16x32_bf16(false, a1f, false, bf, (short)0, acc1, false, false);
    acc2 = __builtin_amdgcn_wmma_f32_16x16x32_bf16(false, a2f, false, bf, (short)0, acc2, false, false);
    acc3 = __builtin_amdgcn_wmma_f32_16x16x32_bf16(false, a3f, false, bf, (short)0, acc3, false, false);
  }

  // ---- epilogue: D layout: lane = column n, VGPR r = row (L<16 ? r : 8+r)
  const float bias = be[e * DOUT + fcol];
  const int mb = (lane < 16) ? 0 : 8;
#pragma unroll
  for (int r = 0; r < 8; ++r) {
    int m = 0 * 16 + mb + r;
    if (m < cnt_local) out[(size_t)srow[m] * DOUT + fcol] = acc0[r] + bias;
    m = 1 * 16 + mb + r;
    if (m < cnt_local) out[(size_t)srow[m] * DOUT + fcol] = acc1[r] + bias;
    m = 2 * 16 + mb + r;
    if (m < cnt_local) out[(size_t)srow[m] * DOUT + fcol] = acc2[r] + bias;
    m = 3 * 16 + mb + r;
    if (m < cnt_local) out[(size_t)srow[m] * DOUT + fcol] = acc3[r] + bias;
  }
}

// ---------------------------------------------------------------- launcher
extern "C" void kernel_launch(void* const* d_in, const int* in_sizes, int n_in,
                              void* d_out, int out_size, void* d_ws, size_t ws_size,
                              hipStream_t stream) {
  const float* X    = (const float*)d_in[0];  // inputs    [N, D_IN]
  const float* cond = (const float*)d_in[1];  // condition [N, D_IN]
  const float* Wp   = (const float*)d_in[2];  // [E, D_IN]
  const float* bp   = (const float*)d_in[3];  // [E]
  const float* We   = (const float*)d_in[4];  // [E, D_OUT, D_IN]
  const float* be   = (const float*)d_in[5];  // [E, D_OUT]
  float* out = (float*)d_out;

  int* counts  = (int*)d_ws;                       // 8 ints
  int* rowlist = (int*)((char*)d_ws + 256);        // E * N ints (256 KB)

  zero_counts_kernel<<<1, 32, 0, stream>>>(counts);
  router_kernel<<<NROWS / 8, 256, 0, stream>>>(cond, Wp, bp, counts, rowlist);

  dim3 grid(DOUT / BN, NEXP * MT);                 // (16, 1024), early-exit on empties
  expert_gemm_kernel<<<grid, 256, 0, stream>>>(X, We, be, counts, rowlist, out);
}
